// MultiHeadAttention_5806795784287
// MI455X (gfx1250) — matrix operand order repaired and checked by reading, not compiled, not run on hardware
//
#include <hip/hip_runtime.h>
#include <hip/hip_bf16.h>

// ---------------------------------------------------------------------------
// MHA forward for MI455X (gfx1250, wave32, WMMA bf16 w/ f32 accum).
// d_out = [ out (B*L*D f32) | p (B*H*L*L f32) ]
// Data movement: TDM tensor_load_to_lds + global_load_async_to_lds for the
// bf16 attention tiles; NT cache hints on the 1 GB probability tensor.
// ---------------------------------------------------------------------------

typedef __attribute__((ext_vector_type(16))) __bf16 bf16x16;
typedef __attribute__((ext_vector_type(8)))  float  floatx8;
typedef float         floatx4 __attribute__((ext_vector_type(4)));
typedef unsigned int  uint32x4 __attribute__((ext_vector_type(4)));
typedef int           int32x4  __attribute__((ext_vector_type(4)));
typedef int           int32x8  __attribute__((ext_vector_type(8)));

static constexpr int Bc = 4, Lc = 2048, Dc = 1024, Hc = 16, DKc = 64;
static constexpr float SCALE = 0.125f;      // 1/sqrt(64)
static constexpr int LDSPITCH = 40;         // 64x32 bf16 tile, padded rows

union FragAB {
    bf16x16 v;
    uint4 q[2];
    unsigned short u[16];
};

__device__ __forceinline__ unsigned short f32_to_bf16(float f) {
    unsigned int u = __float_as_uint(f);
    unsigned int r = (u + 0x7FFFu + ((u >> 16) & 1u)) >> 16;  // RNE
    return (unsigned short)r;
}
__device__ __forceinline__ unsigned int pack2(float a, float b) {
    return (unsigned int)f32_to_bf16(a) | ((unsigned int)f32_to_bf16(b) << 16);
}
__device__ __forceinline__ floatx8 wmma_bf16(const FragAB& a, const FragAB& b, floatx8 c) {
    return __builtin_amdgcn_wmma_f32_16x16x32_bf16(false, a.v, false, b.v,
                                                   (short)0, c, false, false);
}

// Low 32 bits of a generic shared-aperture pointer = LDS byte address.
__device__ __forceinline__ unsigned int lds_addr_of(const void* p) {
    return (unsigned int)(size_t)p;
}

// Async bulk copy: 16B from global to LDS per lane (tracked by ASYNCcnt).
__device__ __forceinline__ void async_copy_b128(void* lds_dst, const void* gsrc) {
    asm volatile("global_load_async_to_lds_b128 %0, %1, off"
                 :: "v"(lds_addr_of(lds_dst)), "v"((unsigned long long)(size_t)gsrc)
                 : "memory");
}
__device__ __forceinline__ void wait_asynccnt0() {
    asm volatile("s_wait_asynccnt 0x0" ::: "memory");
}

// TDM: 1-D tile of `ndwords` dwords, contiguous, global -> LDS.
// D# Group0/Group1 packed per CDNA5 ISA 8.3/8.4. Issued once per wave.
__device__ __forceinline__ void tdm_load_1d(unsigned int lds_addr, const void* gptr,
                                            unsigned int ndwords) {
#if __has_builtin(__builtin_amdgcn_tensor_load_to_lds)
    unsigned long long ga = (unsigned long long)(size_t)gptr;
    uint32x4 g0;
    g0.x = 1u;                                           // count=1 (valid user D#)
    g0.y = lds_addr;                                     // lds_addr [63:32]
    g0.z = (unsigned int)ga;                             // global_addr [95:64]
    g0.w = (unsigned int)((ga >> 32) & 0x01FFFFFFu)      // global_addr hi
         | (2u << 30);                                   // type = 2 ("image")
    int32x8 g1;
    g1[0] = (int)(2u << 16);                             // data_size = 2 -> 4 bytes
    g1[1] = (int)((ndwords & 0xFFFFu) << 16);            // tensor_dim0 lo16 @ [63:48]
    g1[2] = (int)(((ndwords >> 16) & 0xFFFFu)            // tensor_dim0 hi16
         | (1u << 16));                                  // tensor_dim1 = 1 lo16
    g1[3] = (int)((ndwords & 0xFFFFu) << 16);            // tile_dim0 @ [127:112]
    g1[4] = 1;                                           // tile_dim1 = 1
    g1[5] = (int)ndwords;                                // tensor_dim0_stride lo32
    g1[6] = 0;
    g1[7] = 0;
    int32x4 gz = {0, 0, 0, 0};
#if __clang_major__ >= 23
    int32x8 gz8 = {0, 0, 0, 0, 0, 0, 0, 0};
    __builtin_amdgcn_tensor_load_to_lds(g0, g1, gz, gz, gz8, 0);
#else
    __builtin_amdgcn_tensor_load_to_lds(g0, g1, gz, gz, 0);
#endif
#else
    (void)lds_addr; (void)gptr; (void)ndwords;
#endif
}

// ---------------------------------------------------------------------------
// Kernel 1: QKV projections.  Y = X (8192x1024) * W^T (1024x1024)
// z = 0/1/2 selects q/k/v.  Output stored bf16 in per-head layout [B,H,L,64].
// ---------------------------------------------------------------------------
__global__ void __launch_bounds__(256) qkv_proj_kernel(
    const float* __restrict__ qin, const float* __restrict__ kin, const float* __restrict__ vin,
    const float* __restrict__ wq,  const float* __restrict__ wk,  const float* __restrict__ wv,
    unsigned short* __restrict__ qh, unsigned short* __restrict__ kh, unsigned short* __restrict__ vh)
{
    const int m0 = blockIdx.x * 64;
    const int n0 = blockIdx.y * 64;
    const float* X; const float* W; unsigned short* O;
    if (blockIdx.z == 0)      { X = qin; W = wq; O = qh; }
    else if (blockIdx.z == 1) { X = kin; W = wk; O = kh; }
    else                      { X = vin; W = wv; O = vh; }

    __shared__ unsigned short lds_a[64 * LDSPITCH];
    __shared__ unsigned short lds_b[64 * LDSPITCH];

    const int t = threadIdx.x;
    const int wave = t >> 5, lane = t & 31;
    const int mw = (wave & 3) * 16, nw = (wave >> 2) * 32;
    const int lhalf = lane >> 4, l15 = lane & 15;
    const int arow = t >> 2;            // 0..63
    const int acol = (t & 3) * 8;       // 0,8,16,24

    floatx8 acc0 = {0.f,0.f,0.f,0.f,0.f,0.f,0.f,0.f};
    floatx8 acc1 = acc0;

    for (int k0 = 0; k0 < Dc; k0 += 32) {
        if (k0 + 32 < Dc) {
            __builtin_prefetch(&X[(m0 + arow) * Dc + k0 + 32 + acol], 0, 1);
            __builtin_prefetch(&W[(n0 + arow) * Dc + k0 + 32 + acol], 0, 1);
        }
        {   // stage A (64x32 f32 -> bf16)
            const float4* src = (const float4*)&X[(m0 + arow) * Dc + k0 + acol];
            float4 f0 = src[0], f1 = src[1];
            uint4 pk = { pack2(f0.x,f0.y), pack2(f0.z,f0.w), pack2(f1.x,f1.y), pack2(f1.z,f1.w) };
            *(uint4*)&lds_a[arow * LDSPITCH + acol] = pk;
        }
        {   // stage B (64 rows of W x 32 K)
            const float4* src = (const float4*)&W[(n0 + arow) * Dc + k0 + acol];
            float4 f0 = src[0], f1 = src[1];
            uint4 pk = { pack2(f0.x,f0.y), pack2(f0.z,f0.w), pack2(f1.x,f1.y), pack2(f1.z,f1.w) };
            *(uint4*)&lds_b[arow * LDSPITCH + acol] = pk;
        }
        __syncthreads();
        FragAB a, b0, b1;
        a.q[0]  = *(const uint4*)&lds_a[(mw + l15) * LDSPITCH + lhalf * 8];
        a.q[1]  = *(const uint4*)&lds_a[(mw + l15) * LDSPITCH + 16 + lhalf * 8];
        b0.q[0] = *(const uint4*)&lds_b[(nw + l15) * LDSPITCH + lhalf * 8];
        b0.q[1] = *(const uint4*)&lds_b[(nw + l15) * LDSPITCH + 16 + lhalf * 8];
        b1.q[0] = *(const uint4*)&lds_b[(nw + 16 + l15) * LDSPITCH + lhalf * 8];
        b1.q[1] = *(const uint4*)&lds_b[(nw + 16 + l15) * LDSPITCH + 16 + lhalf * 8];
        acc0 = wmma_bf16(a, b0, acc0);
        acc1 = wmma_bf16(a, b1, acc1);
        __syncthreads();
    }

    const int gmbase = m0 + mw + lhalf * 8;
    const int gn0 = n0 + nw + l15;
    const int gn1 = gn0 + 16;
#pragma unroll
    for (int r = 0; r < 8; ++r) {
        int gm = gmbase + r;
        int b_ = gm >> 11, l_ = gm & (Lc - 1);
        O[((b_ * Hc + (gn0 >> 6)) * Lc + l_) * DKc + (gn0 & 63)] = f32_to_bf16(acc0[r]);
        O[((b_ * Hc + (gn1 >> 6)) * Lc + l_) * DKc + (gn1 & 63)] = f32_to_bf16(acc1[r]);
    }
}

// ---------------------------------------------------------------------------
// Kernel 2: attention logits.  S = Qh Kh^T * scale + mask, src_mask fill.
// Q tile (8 KB contiguous bf16) staged via TDM tensor_load_to_lds;
// K tile staged via global_load_async_to_lds_b128. Fragments from LDS.
// ---------------------------------------------------------------------------
__global__ void __launch_bounds__(256) attn_scores_kernel(
    const unsigned short* __restrict__ qh, const unsigned short* __restrict__ kh,
    const float* __restrict__ mask, const int* __restrict__ src_mask,
    float* __restrict__ p)
{
    const int q0  = blockIdx.x * 64;
    const int kk0 = blockIdx.y * 64;
    const int bh  = blockIdx.z;
    const int b_  = bh >> 4;

    __shared__ unsigned short lds_q[64 * 64];   // [m][k], stride 64
    __shared__ unsigned short lds_k[64 * 64];   // [n][k], stride 64

    const int t = threadIdx.x, wave = t >> 5, lane = t & 31;
    const int mw = (wave & 3) * 16, nw = (wave >> 2) * 32;
    const int lhalf = lane >> 4, l15 = lane & 15;

    const unsigned short* qtile = qh + ((size_t)bh * Lc + q0) * DKc;   // 8 KB contiguous
    const unsigned short* ktile = kh + ((size_t)bh * Lc + kk0) * DKc;  // 8 KB contiguous

#if __has_builtin(__builtin_amdgcn_tensor_load_to_lds)
    if (wave == 0)
        tdm_load_1d(lds_addr_of(lds_q), qtile, (64 * 64 * 2) / 4);     // 2048 dwords
#else
    async_copy_b128(lds_q + t * 8,        qtile + t * 8);
    async_copy_b128(lds_q + 2048 + t * 8, qtile + 2048 + t * 8);
#endif
    async_copy_b128(lds_k + t * 8,        ktile + t * 8);
    async_copy_b128(lds_k + 2048 + t * 8, ktile + 2048 + t * 8);

    wait_asynccnt0();
#if __has_builtin(__builtin_amdgcn_tensor_load_to_lds)
    if (wave == 0) __builtin_amdgcn_s_wait_tensorcnt(0);
#endif
    __syncthreads();

    floatx8 acc0 = {0.f,0.f,0.f,0.f,0.f,0.f,0.f,0.f};
    floatx8 acc1 = acc0;

#pragma unroll
    for (int k0 = 0; k0 < DKc; k0 += 32) {
        FragAB a, b0, b1;
        a.q[0]  = *(const uint4*)&lds_q[(mw + l15) * 64 + k0 + lhalf * 8];
        a.q[1]  = *(const uint4*)&lds_q[(mw + l15) * 64 + k0 + 16 + lhalf * 8];
        b0.q[0] = *(const uint4*)&lds_k[(nw + l15) * 64 + k0 + lhalf * 8];
        b0.q[1] = *(const uint4*)&lds_k[(nw + l15) * 64 + k0 + 16 + lhalf * 8];
        b1.q[0] = *(const uint4*)&lds_k[(nw + 16 + l15) * 64 + k0 + lhalf * 8];
        b1.q[1] = *(const uint4*)&lds_k[(nw + 16 + l15) * 64 + k0 + 16 + lhalf * 8];
        acc0 = wmma_bf16(a, b0, acc0);
        acc1 = wmma_bf16(a, b1, acc1);
    }

    const int kn0 = kk0 + nw + l15;
    const int kn1 = kn0 + 16;
    const int sk0 = src_mask[b_ * Lc + kn0];
    const int sk1 = src_mask[b_ * Lc + kn1];
#pragma unroll
    for (int r = 0; r < 8; ++r) {
        int qm = q0 + mw + lhalf * 8 + r;
        int sq = src_mask[b_ * Lc + qm];
        float v0 = acc0[r] * SCALE + mask[((size_t)b_ * Lc + qm) * Lc + kn0];
        float v1 = acc1[r] * SCALE + mask[((size_t)b_ * Lc + qm) * Lc + kn1];
        if ((sq == 0) != (sk0 == 0)) v0 = -1e-9f;   // sm outer product == -1
        if ((sq == 0) != (sk1 == 0)) v1 = -1e-9f;
        size_t prow = ((size_t)bh * Lc + qm) * Lc;
        __builtin_nontemporal_store(v0, &p[prow + kn0]);   // p is streaming (>L2)
        __builtin_nontemporal_store(v1, &p[prow + kn1]);
    }
}

// ---------------------------------------------------------------------------
// Kernel 3: row softmax over p (in place).  One block per row of 2048.
// ---------------------------------------------------------------------------
__global__ void __launch_bounds__(256) softmax_kernel(float* __restrict__ p)
{
    float* rp = p + (size_t)blockIdx.x * Lc;
    floatx4* rp4 = (floatx4*)rp;
    const int t = threadIdx.x;
    __shared__ float red[256];

    floatx4 x0 = __builtin_nontemporal_load(rp4 + t * 2);
    floatx4 x1 = __builtin_nontemporal_load(rp4 + t * 2 + 1);
    float mx = fmaxf(fmaxf(fmaxf(x0.x, x0.y), fmaxf(x0.z, x0.w)),
                     fmaxf(fmaxf(x1.x, x1.y), fmaxf(x1.z, x1.w)));
    red[t] = mx; __syncthreads();
    for (int s = 128; s > 0; s >>= 1) { if (t < s) red[t] = fmaxf(red[t], red[t + s]); __syncthreads(); }
    mx = red[0]; __syncthreads();

    x0.x = __expf(x0.x - mx); x0.y = __expf(x0.y - mx);
    x0.z = __expf(x0.z - mx); x0.w = __expf(x0.w - mx);
    x1.x = __expf(x1.x - mx); x1.y = __expf(x1.y - mx);
    x1.z = __expf(x1.z - mx); x1.w = __expf(x1.w - mx);
    float sum = x0.x + x0.y + x0.z + x0.w + x1.x + x1.y + x1.z + x1.w;
    red[t] = sum; __syncthreads();
    for (int s = 128; s > 0; s >>= 1) { if (t < s) red[t] += red[t + s]; __syncthreads(); }
    float inv = 1.0f / red[0];

    x0 *= inv;
    x1 *= inv;
    __builtin_nontemporal_store(x0, rp4 + t * 2);
    __builtin_nontemporal_store(x1, rp4 + t * 2 + 1);
}

// ---------------------------------------------------------------------------
// Kernel 4: O = P * V.  P (2048x2048 f32 -> bf16 staged), V transposed in LDS.
// Output o stored fp32 as [B, L, H*64].
// ---------------------------------------------------------------------------
__global__ void __launch_bounds__(256) attn_pv_kernel(
    const float* __restrict__ p, const unsigned short* __restrict__ vh,
    float* __restrict__ obuf)
{
    const int q0 = blockIdx.x * 64;
    const int bh = blockIdx.y;
    const int b_ = bh >> 4, h_ = bh & 15;

    __shared__ unsigned short lds_a[64 * LDSPITCH];   // P tile [m][k]
    __shared__ unsigned short lds_b[64 * LDSPITCH];   // V tile transposed [n][k]

    const int t = threadIdx.x, wave = t >> 5, lane = t & 31;
    const int mw = (wave & 3) * 16, nw = (wave >> 2) * 32;
    const int lhalf = lane >> 4, l15 = lane & 15;
    const int arow = t >> 2, acol = (t & 3) * 8;
    const int vk = t >> 3, vn0 = (t & 7) * 8;

    const float* pbase = p + ((size_t)bh * Lc + q0) * Lc;
    const unsigned short* vbase = vh + (size_t)bh * Lc * DKc;

    floatx8 acc0 = {0.f,0.f,0.f,0.f,0.f,0.f,0.f,0.f};
    floatx8 acc1 = acc0;

    for (int k0 = 0; k0 < Lc; k0 += 32) {
        if (k0 + 32 < Lc) {
            __builtin_prefetch(&pbase[(size_t)arow * Lc + k0 + 32 + acol], 0, 1);
            __builtin_prefetch(&vbase[(size_t)(k0 + 32 + vk) * DKc + vn0], 0, 1);
        }
        {   // stage P (64x32 f32 -> bf16), streaming loads
            const floatx4* src = (const floatx4*)&pbase[(size_t)arow * Lc + k0 + acol];
            floatx4 f0 = __builtin_nontemporal_load(src);
            floatx4 f1 = __builtin_nontemporal_load(src + 1);
            uint4 pk = { pack2(f0.x,f0.y), pack2(f0.z,f0.w), pack2(f1.x,f1.y), pack2(f1.z,f1.w) };
            *(uint4*)&lds_a[arow * LDSPITCH + acol] = pk;
        }
        {   // stage V (32x64 bf16), transposed to [n][k]
            union { uint4 q; unsigned short u[8]; } tmp;
            tmp.q = *(const uint4*)&vbase[(size_t)(k0 + vk) * DKc + vn0];
#pragma unroll
            for (int j = 0; j < 8; ++j) lds_b[(vn0 + j) * LDSPITCH + vk] = tmp.u[j];
        }
        __syncthreads();
        FragAB a, b0, b1;
        a.q[0]  = *(const uint4*)&lds_a[(mw + l15) * LDSPITCH + lhalf * 8];
        a.q[1]  = *(const uint4*)&lds_a[(mw + l15) * LDSPITCH + 16 + lhalf * 8];
        b0.q[0] = *(const uint4*)&lds_b[(nw + l15) * LDSPITCH + lhalf * 8];
        b0.q[1] = *(const uint4*)&lds_b[(nw + l15) * LDSPITCH + 16 + lhalf * 8];
        b1.q[0] = *(const uint4*)&lds_b[(nw + 16 + l15) * LDSPITCH + lhalf * 8];
        b1.q[1] = *(const uint4*)&lds_b[(nw + 16 + l15) * LDSPITCH + 16 + lhalf * 8];
        acc0 = wmma_bf16(a, b0, acc0);
        acc1 = wmma_bf16(a, b1, acc1);
        __syncthreads();
    }

    const int n0_ = nw + l15, n1_ = n0_ + 16;
#pragma unroll
    for (int r = 0; r < 8; ++r) {
        int l_ = q0 + mw + lhalf * 8 + r;
        size_t orow = ((size_t)(b_ * Lc + l_)) * (Hc * DKc) + h_ * DKc;
        obuf[orow + n0_] = acc0[r];
        obuf[orow + n1_] = acc1[r];
    }
}

// ---------------------------------------------------------------------------
// Kernel 5: out = O * fc_w^T + residual(q)
// ---------------------------------------------------------------------------
__global__ void __launch_bounds__(256) out_proj_kernel(
    const float* __restrict__ obuf, const float* __restrict__ fc_w,
    const float* __restrict__ resid, float* __restrict__ out)
{
    const int m0 = blockIdx.x * 64;
    const int n0 = blockIdx.y * 64;

    __shared__ unsigned short lds_a[64 * LDSPITCH];
    __shared__ unsigned short lds_b[64 * LDSPITCH];

    const int t = threadIdx.x, wave = t >> 5, lane = t & 31;
    const int mw = (wave & 3) * 16, nw = (wave >> 2) * 32;
    const int lhalf = lane >> 4, l15 = lane & 15;
    const int arow = t >> 2, acol = (t & 3) * 8;

    floatx8 acc0 = {0.f,0.f,0.f,0.f,0.f,0.f,0.f,0.f};
    floatx8 acc1 = acc0;

    for (int k0 = 0; k0 < Dc; k0 += 32) {
        if (k0 + 32 < Dc) {
            __builtin_prefetch(&obuf[(m0 + arow) * Dc + k0 + 32 + acol], 0, 1);
            __builtin_prefetch(&fc_w[(n0 + arow) * Dc + k0 + 32 + acol], 0, 1);
        }
        {
            const float4* src = (const float4*)&obuf[(m0 + arow) * Dc + k0 + acol];
            float4 f0 = src[0], f1 = src[1];
            uint4 pk = { pack2(f0.x,f0.y), pack2(f0.z,f0.w), pack2(f1.x,f1.y), pack2(f1.z,f1.w) };
            *(uint4*)&lds_a[arow * LDSPITCH + acol] = pk;
        }
        {
            const float4* src = (const float4*)&fc_w[(n0 + arow) * Dc + k0 + acol];
            float4 f0 = src[0], f1 = src[1];
            uint4 pk = { pack2(f0.x,f0.y), pack2(f0.z,f0.w), pack2(f1.x,f1.y), pack2(f1.z,f1.w) };
            *(uint4*)&lds_b[arow * LDSPITCH + acol] = pk;
        }
        __syncthreads();
        FragAB a, b0, b1;
        a.q[0]  = *(const uint4*)&lds_a[(mw + l15) * LDSPITCH + lhalf * 8];
        a.q[1]  = *(const uint4*)&lds_a[(mw + l15) * LDSPITCH + 16 + lhalf * 8];
        b0.q[0] = *(const uint4*)&lds_b[(nw + l15) * LDSPITCH + lhalf * 8];
        b0.q[1] = *(const uint4*)&lds_b[(nw + l15) * LDSPITCH + 16 + lhalf * 8];
        b1.q[0] = *(const uint4*)&lds_b[(nw + 16 + l15) * LDSPITCH + lhalf * 8];
        b1.q[1] = *(const uint4*)&lds_b[(nw + 16 + l15) * LDSPITCH + 16 + lhalf * 8];
        acc0 = wmma_bf16(a, b0, acc0);
        acc1 = wmma_bf16(a, b1, acc1);
        __syncthreads();
    }

    const int gn0 = n0 + nw + l15;
    const int gn1 = gn0 + 16;
#pragma unroll
    for (int r = 0; r < 8; ++r) {
        int gm = m0 + mw + lhalf * 8 + r;
        out[(size_t)gm * Dc + gn0] = acc0[r] + resid[(size_t)gm * Dc + gn0];
        out[(size_t)gm * Dc + gn1] = acc1[r] + resid[(size_t)gm * Dc + gn1];
    }
}

// ---------------------------------------------------------------------------
// Kernel 6: LayerNorm over rows of 1024 (in place on out).
// ---------------------------------------------------------------------------
__global__ void __launch_bounds__(256) layernorm_kernel(
    float* __restrict__ out, const float* __restrict__ g, const float* __restrict__ bta)
{
    float* rp = out + (size_t)blockIdx.x * Dc;
    const int t = threadIdx.x;
    __shared__ float rs[256];
    __shared__ float rs2[256];

    float4 x = ((float4*)rp)[t];
    float s  = x.x + x.y + x.z + x.w;
    float s2 = x.x*x.x + x.y*x.y + x.z*x.z + x.w*x.w;
    rs[t] = s; rs2[t] = s2; __syncthreads();
    for (int k = 128; k > 0; k >>= 1) {
        if (t < k) { rs[t] += rs[t + k]; rs2[t] += rs2[t + k]; }
        __syncthreads();
    }
    const float mean = rs[0] * (1.0f / Dc);
    const float var  = rs2[0] * (1.0f / Dc) - mean * mean;
    const float inv  = rsqrtf(var + 1e-6f);

    const float4 gg = ((const float4*)g)[t];
    const float4 bb = ((const float4*)bta)[t];
    x.x = (x.x - mean) * inv * gg.x + bb.x;
    x.y = (x.y - mean) * inv * gg.y + bb.y;
    x.z = (x.z - mean) * inv * gg.z + bb.z;
    x.w = (x.w - mean) * inv * gg.w + bb.w;
    ((float4*)rp)[t] = x;
}

// ---------------------------------------------------------------------------
extern "C" void kernel_launch(void* const* d_in, const int* in_sizes, int n_in,
                              void* d_out, int out_size, void* d_ws, size_t ws_size,
                              hipStream_t stream) {
    (void)in_sizes; (void)n_in; (void)out_size; (void)ws_size;
    const float* q        = (const float*)d_in[0];
    const float* k        = (const float*)d_in[1];
    const float* v        = (const float*)d_in[2];
    const float* mask     = (const float*)d_in[3];
    const int*   src_mask = (const int*)  d_in[4];
    const float* wq       = (const float*)d_in[5];
    const float* wk       = (const float*)d_in[6];
    const float* wv       = (const float*)d_in[7];
    const float* fc_w     = (const float*)d_in[8];
    const float* ln_g     = (const float*)d_in[9];
    const float* ln_b     = (const float*)d_in[10];

    float* out = (float*)d_out;                       // B*L*D
    float* p   = out + (size_t)Bc * Lc * Dc;          // B*H*L*L

    const size_t headElems = (size_t)Bc * Hc * Lc * DKc;   // 8.4M
    unsigned short* qh = (unsigned short*)d_ws;
    unsigned short* kh = qh + headElems;
    unsigned short* vh = kh + headElems;
    float* obuf = (float*)(vh + headElems);                // B*L*H*DV f32

    dim3 g1(Bc * Lc / 64, Dc / 64, 3);
    qkv_proj_kernel<<<g1, 256, 0, stream>>>(q, k, v, wq, wk, wv, qh, kh, vh);

    dim3 g2(Lc / 64, Lc / 64, Bc * Hc);
    attn_scores_kernel<<<g2, 256, 0, stream>>>(qh, kh, mask, src_mask, p);

    softmax_kernel<<<dim3(Bc * Hc * Lc), 256, 0, stream>>>(p);

    dim3 g4(Lc / 64, Bc * Hc);
    attn_pv_kernel<<<g4, 256, 0, stream>>>(p, vh, obuf);

    out_proj_kernel<<<dim3(Bc * Lc / 64, Dc / 64), 256, 0, stream>>>(obuf, fc_w, q, out);

    layernorm_kernel<<<dim3(Bc * Lc), 256, 0, stream>>>(out, ln_g, ln_b);
}
